// Agent_57294863728980
// MI455X (gfx1250) — matrix operand order repaired and checked by reading, not compiled, not run on hardware
//
#include <hip/hip_runtime.h>

// ---------------------------------------------------------------------------
// CDNA5 (gfx1250) implementation of the IMPALA-style agent forward pass.
// Every layer runs through one bf16 WMMA GEMM kernel (v_wmma_f32_16x16x32_bf16,
// wave32). Weights are cast f32->bf16 (with layout permutations folded in)
// once per launch into the workspace.
//
// Round-4: K loop manually unrolled 2x with two alternating fragment sets so
// software pipelining needs no register rotation (kills the v_dual_mov storm
// from round 3). Loads for one set issue while WMMAs consume the other set;
// waits stay partial (s_wait_loadcnt > 0). Control flow is scalar throughout
// (readfirstlane'd wave id) -> EXEC stays all-ones around every WMMA.
// Requires K % 64 == 0 (true for all layers: 64,4096,512,7680,2048,1024).
// ---------------------------------------------------------------------------

typedef __bf16 bf16;
typedef __attribute__((ext_vector_type(16))) __bf16 v16bf;
typedef __attribute__((ext_vector_type(8)))  __bf16 v8bf;
typedef __attribute__((ext_vector_type(8)))  float  v8f;

__device__ __forceinline__ bf16 f2bf(float f) {
  unsigned int u = __builtin_bit_cast(unsigned int, f);
  unsigned int r = u + 0x7FFFu + ((u >> 16) & 1u);   // round-to-nearest-even
  return __builtin_bit_cast(bf16, (unsigned short)(r >> 16));
}
__device__ __forceinline__ float bf2f(bf16 b) {
  unsigned int u = ((unsigned int)__builtin_bit_cast(unsigned short, b)) << 16;
  return __builtin_bit_cast(float, u);
}
__device__ __forceinline__ float leakyf(float v) { return v > 0.f ? v : 0.01f * v; }
__device__ __forceinline__ float sigmf(float v)  { return 1.f / (1.f + expf(-v)); }

// ---------------------------------------------------------------------------
// WMMA GEMM: C[M,N] = A[M,K] (bf16 row-major) x W[N,K]^T (bf16 torch layout)
// + bias, optional LeakyReLU; writes f32 and/or bf16. M,N multiples of 16,
// K multiple of 64, (M/16) a multiple of MT.
// One wave owns one 16-wide column strip x MT row tiles.
// ---------------------------------------------------------------------------
template <int MT>
__global__ void __launch_bounds__(256)
k_wmma_gemm(const bf16* __restrict__ A, const bf16* __restrict__ W,
            const float* __restrict__ bias, int nbias,
            float* __restrict__ Cf, bf16* __restrict__ Cb,
            int M, int N, int K, int leaky)
{
  const int lane = threadIdx.x & 31;
  // readfirstlane -> SGPR: all wave-level control flow stays scalar.
  const int wv   = __builtin_amdgcn_readfirstlane((int)(threadIdx.x >> 5));
  const int wid  = (blockIdx.x << 3) + wv;
  const int ntiles  = N >> 4;
  const int mgroups = (M >> 4) / MT;
  if (wid >= ntiles * mgroups) return;            // scalar branch
  const int nt = wid % ntiles;
  const int m0 = (wid / ntiles) * MT;

  const int ln = lane & 15;                       // tile column / A-frag row
  const int lh = lane >> 4;                       // half-wave K split

  // B fragment (32x16): lane holds N=nt*16+ln, K = kb + lh*16 .. +15 (contig)
  const bf16* wp  = W + (size_t)(nt * 16 + ln) * K + (lh << 4);
  // A fragment (16x32): lane holds M=m*16+ln; K=kb+lh*16..+15 (contig)
  const bf16* ap0 = A + (size_t)(m0 * 16 + ln) * K + (lh << 4);
  const size_t astride = (size_t)16 * K;

  auto loadB = [&](int kb) -> v16bf {
    return *(const v16bf*)(wp + kb);
  };
  auto loadA = [&](int i, int kb) -> v16bf {
    const bf16* ap = ap0 + (size_t)i * astride + kb;
    v8bf alo = *(const v8bf*)(ap);
    v8bf ahi = *(const v8bf*)(ap + 8);
    return __builtin_shufflevector(alo, ahi,
        0,1,2,3,4,5,6,7,8,9,10,11,12,13,14,15);
  };

  v8f acc[MT];
#pragma unroll
  for (int i = 0; i < MT; ++i) acc[i] = {};

  // ---- 2x-unrolled, double-buffered K loop (no register rotation) ----
  v16bf bX = loadB(0);                 // set X holds block kb
  v16bf aX[MT];
#pragma unroll
  for (int i = 0; i < MT; ++i) aX[i] = loadA(i, 0);

  for (int kb = 0; kb < K; kb += 64) {
    // issue set Y loads (block kb+32) before consuming set X
    v16bf bY = loadB(kb + 32);
    v16bf aY[MT];
#pragma unroll
    for (int i = 0; i < MT; ++i) aY[i] = loadA(i, kb + 32);
    __builtin_prefetch(wp + kb + 64, 0, 3);       // global_prefetch_b8
#pragma unroll
    for (int i = 0; i < MT; ++i)
      acc[i] = __builtin_amdgcn_wmma_f32_16x16x32_bf16(
          false, aX[i], false, bX, (short)0, acc[i], false, false);
    // issue set X loads (block kb+64) before consuming set Y
    if (kb + 64 < K) {                            // scalar-uniform branch
      bX = loadB(kb + 64);
#pragma unroll
      for (int i = 0; i < MT; ++i) aX[i] = loadA(i, kb + 64);
    }
#pragma unroll
    for (int i = 0; i < MT; ++i)
      acc[i] = __builtin_amdgcn_wmma_f32_16x16x32_bf16(
          false, aY[i], false, bY, (short)0, acc[i], false, false);
  }

  // C/D layout: lane ln = column, VGPR r holds row r + lh*8 within tile
  const int col = nt * 16 + ln;
  const float bv = (bias != nullptr && col < nbias) ? bias[col] : 0.f;
#pragma unroll
  for (int i = 0; i < MT; ++i) {
#pragma unroll
    for (int r = 0; r < 8; ++r) {
      int row = (m0 + i) * 16 + (lh << 3) + r;
      float v = acc[i][r] + bv;
      if (leaky) v = leakyf(v);
      size_t o = (size_t)row * N + col;
      if (Cf) Cf[o] = v;
      if (Cb) Cb[o] = f2bf(v);
    }
  }
}

// ------------------------- weight / bias packing ---------------------------

__global__ void k_cvt_pad_rows(const float* __restrict__ src, bf16* __restrict__ dst,
                               int Ns, int K, int total /* = Nd*K */)
{
  int i = blockIdx.x * blockDim.x + threadIdx.x;
  if (i >= total) return;
  int row = i / K;
  dst[i] = (row < Ns) ? f2bf(src[i]) : __builtin_bit_cast(bf16, (unsigned short)0);
}

// OIHW conv weight -> [O, (kh*KW+kw)*Cin + c] bf16 (matches NHWC im2col order)
__global__ void k_pack_conv_w(const float* __restrict__ src, bf16* __restrict__ dst,
                              int Cin, int KH, int KW, int total)
{
  int i = blockIdx.x * blockDim.x + threadIdx.x;
  if (i >= total) return;
  int K = Cin * KH * KW;
  int o = i / K, r = i % K;
  int c = r % Cin, s = r / Cin;
  int kw = s % KW, kh = s / KW;
  dst[i] = f2bf(src[(((size_t)o * Cin + c) * KH + kh) * KW + kw]);
}

// fc1 weight: k_ref = c*60 + s  ->  k_ours = s*128 + c   (s = oh*12+ow)
__global__ void k_pack_fc1_w(const float* __restrict__ src, bf16* __restrict__ dst,
                             int total /* 4096*7680 */)
{
  int i = blockIdx.x * blockDim.x + threadIdx.x;
  if (i >= total) return;
  int o = i / 7680, r = i % 7680;
  int c = r & 127, s = r >> 7;
  dst[i] = f2bf(src[(size_t)o * 7680 + c * 60 + s]);
}

__global__ void k_bias_comb(const float* a, const float* b, float* out, int n)
{
  int i = blockIdx.x * blockDim.x + threadIdx.x;
  if (i < n) out[i] = a[i] + b[i];
}

// ----------------------------- im2col / pool -------------------------------

// conv1 im2col: A1[(n*33+oh)*61+ow][kh*8+kw] = x[n, oh*2+kh, ow*2+kw, 0]/255
__global__ void k_im2col1(const float* __restrict__ x, bf16* __restrict__ A1, int total)
{
  int i = blockIdx.x * blockDim.x + threadIdx.x;
  if (i >= total) return;
  int k = i & 63, m = i >> 6;
  int ow = m % 61, t = m / 61;
  int oh = t % 33, n = t / 33;
  int kh = k >> 3, kw = k & 7;
  float v = x[((size_t)n * 72 + oh * 2 + kh) * 128 + ow * 2 + kw] * (1.f / 255.f);
  A1[i] = f2bf(v);
}

// MaxPool2d(4, stride 2) over conv1 output (NHWC bf16), then LeakyReLU
__global__ void k_pool_leaky(const bf16* __restrict__ C1, bf16* __restrict__ P, int total)
{
  int i = blockIdx.x * blockDim.x + threadIdx.x;
  if (i >= total) return;
  int c = i & 255, t = i >> 8;
  int pw = t % 29; t /= 29;
  int ph = t % 15; int n = t / 15;
  float m = -3.4e38f;
#pragma unroll
  for (int a = 0; a < 4; ++a)
#pragma unroll
    for (int b = 0; b < 4; ++b) {
      float v = bf2f(C1[(((size_t)n * 33 + ph * 2 + a) * 61 + (pw * 2 + b)) * 256 + c]);
      m = v > m ? v : m;
    }
  P[i] = f2bf(leakyf(m));
}

// conv2 im2col: A2[(n*6+oh)*13+ow][(kh*4+kw)*256+c] = P[n, oh*2+kh, ow*2+kw, c]
__global__ void k_im2col2(const bf16* __restrict__ P, bf16* __restrict__ A2, int total)
{
  int i = blockIdx.x * blockDim.x + threadIdx.x;
  if (i >= total) return;
  int k = i & 4095, m = i >> 12;
  int c = k & 255, s = k >> 8;
  int kw = s & 3, kh = s >> 2;
  int ow = m % 13, t = m / 13;
  int oh = t % 6, n = t / 6;
  A2[i] = P[(((size_t)n * 15 + oh * 2 + kh) * 29 + (ow * 2 + kw)) * 256 + c];
}

// conv3 im2col: A3[(n*5+oh)*12+ow][(kh*2+kw)*128+c] = C2[n, oh+kh, ow+kw, c]
__global__ void k_im2col3(const bf16* __restrict__ C2, bf16* __restrict__ A3, int total)
{
  int i = blockIdx.x * blockDim.x + threadIdx.x;
  if (i >= total) return;
  int k = i & 511, m = i >> 9;
  int c = k & 127, s = k >> 7;
  int kw = s & 1, kh = s >> 1;
  int ow = m % 12, t = m / 12;
  int oh = t % 5, n = t / 5;
  A3[i] = C2[(((size_t)n * 6 + oh + kh) * 13 + (ow + kw)) * 128 + c];
}

// ------------------------------- LSTM bits ---------------------------------

__global__ void k_init_state(const float* h0, const float* c0,
                             float* hS, float* cS, bf16* hb /* [16,2048] */)
{
  int i = blockIdx.x * blockDim.x + threadIdx.x;
  if (i >= 16 * 2048) return;
  hb[i] = __builtin_bit_cast(bf16, (unsigned short)0);   // rows 4..15 stay zero
  if (i < 4 * 2048) { hS[i] = h0[i]; cS[i] = c0[i]; }
}

// hb[b] = bf16(h * (1 - done[t*4+b]))  -- done-mask applied before Whh matmul
__global__ void k_mask_h(const float* __restrict__ done, const float* __restrict__ hS,
                         bf16* __restrict__ hb, int t)
{
  int i = blockIdx.x * blockDim.x + threadIdx.x;
  if (i >= 4 * 2048) return;
  int b = i >> 11;
  float m = 1.f - done[t * 4 + b];
  hb[i] = f2bf(hS[i] * m);
}

// gates = Gx[t*4+b] + Gh[b] + (bih+bhh);  i,f,g,o order; H = 2048
__global__ void k_lstm_cell(const float* __restrict__ Gx, const float* __restrict__ Gh,
                            const float* __restrict__ biasc, const float* __restrict__ done,
                            float* __restrict__ hS, float* __restrict__ cS,
                            bf16* __restrict__ hid, int t)
{
  int i = blockIdx.x * blockDim.x + threadIdx.x;
  if (i >= 4 * 2048) return;
  int b = i >> 11, j = i & 2047;
  int row = t * 4 + b;
  size_t gx = (size_t)row * 8192, gh = (size_t)b * 8192;
  float gi = Gx[gx + j]        + Gh[gh + j]        + biasc[j];
  float gf = Gx[gx + 2048 + j] + Gh[gh + 2048 + j] + biasc[2048 + j];
  float gg = Gx[gx + 4096 + j] + Gh[gh + 4096 + j] + biasc[4096 + j];
  float go = Gx[gx + 6144 + j] + Gh[gh + 6144 + j] + biasc[6144 + j];
  float m  = 1.f - done[row];
  float c  = cS[i] * m;
  float cn = sigmf(gf) * c + sigmf(gi) * tanhf(gg);
  float hn = sigmf(go) * tanhf(cn);
  cS[i] = cn; hS[i] = hn;
  hid[(size_t)row * 2048 + j] = f2bf(hn);
}

// ------------------------------- finalize ----------------------------------
// d_out = [logits 128x36][value 128x1][hn 1x4x2048][cn 1x4x2048]
__global__ void k_finalize(const float* __restrict__ Ca3, const float* __restrict__ Cv3,
                           const float* __restrict__ hS, const float* __restrict__ cS,
                           float* __restrict__ out)
{
  int i = blockIdx.x * blockDim.x + threadIdx.x;
  if (i >= 21120) return;
  if (i < 4608)          { int n = i / 36, j = i % 36; out[i] = Ca3[n * 48 + j]; }
  else if (i < 4736)     { int n = i - 4608;           out[i] = Cv3[n * 16]; }
  else if (i < 12928)    { out[i] = hS[i - 4736]; }
  else                   { out[i] = cS[i - 12928]; }
}

// ---------------------------------------------------------------------------

static inline int cdiv(int a, int b) { return (a + b - 1) / b; }

extern "C" void kernel_launch(void* const* d_in, const int* in_sizes, int n_in,
                              void* d_out, int out_size, void* d_ws, size_t ws_size,
                              hipStream_t stream)
{
  const float* x    = (const float*)d_in[0];
  const float* done = (const float*)d_in[1];
  const float* h0   = (const float*)d_in[2];
  const float* c0   = (const float*)d_in[3];
  const float* c1w  = (const float*)d_in[4];  const float* c1b = (const float*)d_in[5];
  const float* c2w  = (const float*)d_in[6];  const float* c2b = (const float*)d_in[7];
  const float* c3w  = (const float*)d_in[8];  const float* c3b = (const float*)d_in[9];
  const float* f1w  = (const float*)d_in[10]; const float* f1b = (const float*)d_in[11];
  const float* f2w  = (const float*)d_in[12]; const float* f2b = (const float*)d_in[13];
  const float* wih  = (const float*)d_in[14]; const float* whh = (const float*)d_in[15];
  const float* bih  = (const float*)d_in[16]; const float* bhh = (const float*)d_in[17];
  const float* a1w  = (const float*)d_in[18]; const float* a1b = (const float*)d_in[19];
  const float* a2w  = (const float*)d_in[20]; const float* a2b = (const float*)d_in[21];
  const float* a3w  = (const float*)d_in[22]; const float* a3b = (const float*)d_in[23];
  const float* v1w  = (const float*)d_in[24]; const float* v1b = (const float*)d_in[25];
  const float* v2w  = (const float*)d_in[26]; const float* v2b = (const float*)d_in[27];
  const float* v3w  = (const float*)d_in[28]; const float* v3b = (const float*)d_in[29];
  float* out = (float*)d_out;

  // ---- workspace bump allocator (256B aligned) ----
  char* p = (char*)d_ws;
  auto alloc = [&](size_t bytes) -> void* {
    void* r = (void*)p; p += (bytes + 255) & ~(size_t)255; return r;
  };
  bf16*  Wc1  = (bf16*) alloc((size_t)256 * 64 * 2);
  bf16*  Wc2  = (bf16*) alloc((size_t)128 * 4096 * 2);
  bf16*  Wc3  = (bf16*) alloc((size_t)128 * 512 * 2);
  bf16*  Wf1  = (bf16*) alloc((size_t)4096 * 7680 * 2);
  bf16*  Wf2  = (bf16*) alloc((size_t)4096 * 4096 * 2);
  bf16*  Wih  = (bf16*) alloc((size_t)8192 * 4096 * 2);
  bf16*  Whh  = (bf16*) alloc((size_t)8192 * 2048 * 2);
  bf16*  Wa1  = (bf16*) alloc((size_t)1024 * 2048 * 2);
  bf16*  Wa2  = (bf16*) alloc((size_t)512 * 1024 * 2);
  bf16*  Wa3  = (bf16*) alloc((size_t)48 * 512 * 2);
  bf16*  Wv1  = (bf16*) alloc((size_t)1024 * 2048 * 2);
  bf16*  Wv2  = (bf16*) alloc((size_t)512 * 1024 * 2);
  bf16*  Wv3  = (bf16*) alloc((size_t)16 * 512 * 2);
  float* Bc   = (float*)alloc((size_t)8192 * 4);
  bf16*  A1   = (bf16*) alloc((size_t)257664 * 64 * 2);
  bf16*  C1   = (bf16*) alloc((size_t)257664 * 256 * 2);
  bf16*  P    = (bf16*) alloc((size_t)14254080 * 2);
  bf16*  A2   = (bf16*) alloc((size_t)9984 * 4096 * 2);
  bf16*  C2   = (bf16*) alloc((size_t)9984 * 128 * 2);
  bf16*  A3   = (bf16*) alloc((size_t)7680 * 512 * 2);
  bf16*  C3   = (bf16*) alloc((size_t)7680 * 128 * 2);   // == fc1 A [128,7680]
  bf16*  F1   = (bf16*) alloc((size_t)128 * 4096 * 2);
  bf16*  F2   = (bf16*) alloc((size_t)128 * 4096 * 2);
  float* Gx   = (float*)alloc((size_t)128 * 8192 * 4);
  bf16*  Hb   = (bf16*) alloc((size_t)16 * 2048 * 2);
  float* Gh   = (float*)alloc((size_t)16 * 8192 * 4);
  float* HS   = (float*)alloc((size_t)4 * 2048 * 4);
  float* CS   = (float*)alloc((size_t)4 * 2048 * 4);
  bf16*  HID  = (bf16*) alloc((size_t)128 * 2048 * 2);
  bf16*  Ao1  = (bf16*) alloc((size_t)128 * 1024 * 2);
  bf16*  Ao2  = (bf16*) alloc((size_t)128 * 512 * 2);
  float* Ca3  = (float*)alloc((size_t)128 * 48 * 4);
  bf16*  Vo1  = (bf16*) alloc((size_t)128 * 1024 * 2);
  bf16*  Vo2  = (bf16*) alloc((size_t)128 * 512 * 2);
  float* Cv3  = (float*)alloc((size_t)128 * 16 * 4);

  // MT chosen so it divides (M/16) exactly -> no guards inside the kernel.
  auto gemm = [&](const bf16* A, const bf16* W, const float* bias, int nbias,
                  float* Cf, bf16* Cb, int M, int N, int K, int leaky) {
    int mtiles = M / 16;
    if ((mtiles & 3) == 0) {
      int nwork = (N / 16) * (mtiles / 4);
      k_wmma_gemm<4><<<dim3(cdiv(nwork, 8)), dim3(256), 0, stream>>>(
          A, W, bias, nbias, Cf, Cb, M, N, K, leaky);
    } else {
      int nwork = (N / 16) * mtiles;
      k_wmma_gemm<1><<<dim3(cdiv(nwork, 8)), dim3(256), 0, stream>>>(
          A, W, bias, nbias, Cf, Cb, M, N, K, leaky);
    }
  };

  // ---- weight packing (f32 -> bf16, permuted) ----
  {
    int t;
    t = 256 * 64;       k_cvt_pad_rows<<<cdiv(t,256),256,0,stream>>>(c1w, Wc1, 256, 64, t);
    t = 128 * 4096;     k_pack_conv_w <<<cdiv(t,256),256,0,stream>>>(c2w, Wc2, 256, 4, 4, t);
    t = 128 * 512;      k_pack_conv_w <<<cdiv(t,256),256,0,stream>>>(c3w, Wc3, 128, 2, 2, t);
    t = 4096 * 7680;    k_pack_fc1_w  <<<cdiv(t,256),256,0,stream>>>(f1w, Wf1, t);
    t = 4096 * 4096;    k_cvt_pad_rows<<<cdiv(t,256),256,0,stream>>>(f2w, Wf2, 4096, 4096, t);
    t = 8192 * 4096;    k_cvt_pad_rows<<<cdiv(t,256),256,0,stream>>>(wih, Wih, 8192, 4096, t);
    t = 8192 * 2048;    k_cvt_pad_rows<<<cdiv(t,256),256,0,stream>>>(whh, Whh, 8192, 2048, t);
    t = 1024 * 2048;    k_cvt_pad_rows<<<cdiv(t,256),256,0,stream>>>(a1w, Wa1, 1024, 2048, t);
    t = 512 * 1024;     k_cvt_pad_rows<<<cdiv(t,256),256,0,stream>>>(a2w, Wa2, 512, 1024, t);
    t = 48 * 512;       k_cvt_pad_rows<<<cdiv(t,256),256,0,stream>>>(a3w, Wa3, 36, 512, t);
    t = 1024 * 2048;    k_cvt_pad_rows<<<cdiv(t,256),256,0,stream>>>(v1w, Wv1, 1024, 2048, t);
    t = 512 * 1024;     k_cvt_pad_rows<<<cdiv(t,256),256,0,stream>>>(v2w, Wv2, 512, 1024, t);
    t = 16 * 512;       k_cvt_pad_rows<<<cdiv(t,256),256,0,stream>>>(v3w, Wv3, 1, 512, t);
    k_bias_comb<<<cdiv(8192,256),256,0,stream>>>(bih, bhh, Bc, 8192);
  }

  // ---- conv backbone (im2col + WMMA GEMM) ----
  int t1 = 257664 * 64;
  k_im2col1<<<cdiv(t1,256),256,0,stream>>>(x, A1, t1);
  gemm(A1, Wc1, c1b, 256, nullptr, C1, 257664, 256, 64, 0);      // conv1 (pre-pool)
  int tp = 14254080;
  k_pool_leaky<<<cdiv(tp,256),256,0,stream>>>(C1, P, tp);        // maxpool(4,2)+leaky
  int t2 = 9984 * 4096;
  k_im2col2<<<cdiv(t2,256),256,0,stream>>>(P, A2, t2);
  gemm(A2, Wc2, c2b, 128, nullptr, C2, 9984, 128, 4096, 1);      // conv2 + leaky
  int t3 = 7680 * 512;
  k_im2col3<<<cdiv(t3,256),256,0,stream>>>(C2, A3, t3);
  gemm(A3, Wc3, c3b, 128, nullptr, C3, 7680, 128, 512, 1);       // conv3 + leaky

  // ---- FC trunk (C3 is byte-identical to the [128,7680] flatten) ----
  gemm(C3, Wf1, f1b, 4096, nullptr, F1, 128, 4096, 7680, 1);     // fc1 + leaky
  gemm(F1, Wf2, f2b, 4096, nullptr, F2, 128, 4096, 4096, 1);     // fc2 + leaky

  // ---- LSTM: batched x*Wih^T, then 32 sequential h*Whh^T + cell steps ----
  gemm(F2, Wih, nullptr, 0, Gx, nullptr, 128, 8192, 4096, 0);    // all timesteps at once
  k_init_state<<<cdiv(16*2048,256),256,0,stream>>>(h0, c0, HS, CS, Hb);
  for (int t = 0; t < 32; ++t) {
    k_mask_h<<<cdiv(4*2048,256),256,0,stream>>>(done, HS, Hb, t);
    gemm(Hb, Whh, nullptr, 0, Gh, nullptr, 16, 8192, 2048, 0);   // M padded 4->16, MT=1
    k_lstm_cell<<<cdiv(4*2048,256),256,0,stream>>>(Gx, Gh, Bc, done, HS, CS, HID, t);
  }

  // ---- heads ----
  gemm(HID, Wa1, a1b, 1024, nullptr, Ao1, 128, 1024, 2048, 1);
  gemm(Ao1, Wa2, a2b, 512,  nullptr, Ao2, 128, 512, 1024, 1);
  gemm(Ao2, Wa3, a3b, 36,   Ca3, nullptr, 128, 48, 512, 0);      // logits (N pad 36->48)
  gemm(HID, Wv1, v1b, 1024, nullptr, Vo1, 128, 1024, 2048, 1);
  gemm(Vo1, Wv2, v2b, 512,  nullptr, Vo2, 128, 512, 1024, 1);
  gemm(Vo2, Wv3, v3b, 1,    Cv3, nullptr, 128, 16, 512, 0);      // value (N pad 1->16)

  k_finalize<<<cdiv(21120,256),256,0,stream>>>(Ca3, Cv3, HS, CS, out);
}
